// NanoLLM_9174050144318
// MI455X (gfx1250) — matrix operand order repaired and checked by reading, not compiled, not run on hardware
//
#include <hip/hip_runtime.h>
#include <hip/hip_bf16.h>
#include <math.h>

typedef __attribute__((ext_vector_type(16))) _Float16 v16h;
typedef __attribute__((ext_vector_type(8)))  _Float16 v8h;
typedef __attribute__((ext_vector_type(8)))  float    v8f;

// ---------------- problem constants (from setup_inputs) ----------------
#define BSZ    4
#define LSEQ   2048
#define MROWS  (BSZ * LSEQ)      // 8192
#define EMB    512
#define DMODEL 768
#define ED     768
#define N2     (2 * ED)          // 1536
#define VOCAB  6400
#define DTRANK 48
#define DBCW   64                // dt_rank + 2N = 50, padded to 64
#define NLAYER 8

// =====================================================================
// CDNA5 async Global->LDS copy (ASYNCcnt-tracked, ISA ch.10 / §15.18).
// dst32 = LDS byte address (low 32 bits of a generic shared pointer,
// per the aperture rule); src = 64-bit VGPR pair.
// =====================================================================
__device__ __forceinline__ void async_b128_to_lds(unsigned int dst32, const void* src) {
    asm volatile("global_load_async_to_lds_b128 %0, %1, off"
                 :: "v"(dst32), "v"(src) : "memory");
}
__device__ __forceinline__ void wait_async0(void) {
    asm volatile("s_wait_asynccnt 0" ::: "memory");
}
__device__ __forceinline__ void wait_async1(void) {
    asm volatile("s_wait_asynccnt 1" ::: "memory");
}

// A fragment 16x32 (f16): elems 0..15 = K(half*16+i)
__device__ __forceinline__ v16h load_a_frag(const _Float16* p) {
    v8h lo = *(const v8h*)(p);
    v8h hi = *(const v8h*)(p + 8);
    v16h a;
    #pragma unroll
    for (int i = 0; i < 8; ++i) { a[i] = lo[i]; a[i + 8] = hi[i]; }
    return a;
}

// =====================================================================
// Weight transpose + f32 -> f16 convert:  W[K][N] -> Wt[Np][Kp] (zero pad)
// =====================================================================
__global__ void k_transpose_w(const float* __restrict__ W, _Float16* __restrict__ Wt,
                              int K, int N, int Kp, int Np) {
    int t = blockIdx.x * 256 + threadIdx.x;
    if (t >= Np * Kp) return;
    int n = t / Kp, k = t % Kp;
    _Float16 v = (_Float16)0.0f;
    if (n < N && k < K) v = (_Float16)W[(size_t)k * N + n];
    Wt[t] = v;
}

// =====================================================================
// Embedding gather -> f16 activation matrix (MROWS x EMB)
// =====================================================================
__global__ void k_embed_gather(const int* __restrict__ tok,
                               const float* __restrict__ emb,
                               _Float16* __restrict__ out) {
    int t = blockIdx.x * 256 + threadIdx.x;         // < MROWS*EMB
    int m = t >> 9, k = t & 511;                    // EMB == 512
    out[t] = (_Float16)emb[(size_t)tok[m] * EMB + k];
}

// =====================================================================
// WMMA GEMM:  C[M,N] = A[M,K](f16,row-major) * Bt[N,K](f16, N-major) (+bias)
// grid = (N/64, M/256), block = 256 (8 waves); wave tile = 32x64 (8 wmma/step).
// B tile (64x32 f16 = 4KB) staged in LDS via async Global->LDS copy, double
// buffered, shared by all 8 waves. A fragments register-double-buffered from
// global. Steady-state loop is branchless (last stage peeled).
// A/B fragments use contiguous k = half*16 + i (wave32); C/D as in the ISA.
// =====================================================================
__global__ __launch_bounds__(256) void k_gemm_f16(
    const _Float16* __restrict__ A, int lda,
    const _Float16* __restrict__ Bt, int ldb,
    float* __restrict__ C, int ldc,
    const float* __restrict__ bias, int K)
{
    __shared__ _Float16 sB[2][64 * 32];   // [buf][col*32 + k], 8 KB total

    const int tid  = threadIdx.x;
    const int lane = tid & 31;
    const int wave = tid >> 5;
    const int half = lane >> 4;           // K half-select
    const int m16  = lane & 15;           // A: row in tile / B,C: column
    const int row0 = blockIdx.y * 256 + wave * 32;
    const int col0 = blockIdx.x * 64;

    // cooperative B-tile copy: thread moves one 16B chunk (256*16B = 4KB)
    const int cc = tid >> 2;              // column 0..63
    const int ch = tid & 3;               // 16B chunk 0..3 within 64B row
    const _Float16*    bsrc  = Bt + (size_t)(col0 + cc) * ldb + ch * 8;
    const unsigned int dst0  = (unsigned int)(uintptr_t)&sB[0][cc * 32 + ch * 8];
    const unsigned int dst1  = dst0 + 64 * 32 * 2;     // buffer 1 (+4096 B)

    v8f acc[2][4] = {};

    const _Float16* ap0 = A + (size_t)(row0 +      m16) * lda + half * 16;
    const _Float16* ap1 = A + (size_t)(row0 + 16 + m16) * lda + half * 16;

    // prologue: stage 0 B-tile into buffer 0; stage 0 A fragments into regs
    async_b128_to_lds(dst0, bsrc);
    v16h a0c = load_a_frag(ap0);
    v16h a1c = load_a_frag(ap1);

    const int S = K >> 5;                 // K stages of 32
    for (int s = 0; s < S - 1; ++s) {
        // software pipeline: next-stage A fragments + async B copy first
        v16h a0n = load_a_frag(ap0 + 32);
        v16h a1n = load_a_frag(ap1 + 32);
        async_b128_to_lds(((s + 1) & 1) ? dst1 : dst0, bsrc + (size_t)(s + 1) * 32);
        wait_async1();                    // stage s B landed (in-order completion)
        __syncthreads();                  // tile visible to all 8 waves

        const _Float16* lb = &sB[s & 1][half * 16];
        #pragma unroll
        for (int j = 0; j < 4; ++j) {     // B frag 32x16: lane=N col, elems=K(half*16+i)
            v16h b = *(const v16h*)(lb + (j * 16 + m16) * 32);
            acc[0][j] = __builtin_amdgcn_wmma_f32_16x16x32_f16(false, a0c, false, b, (short)0, acc[0][j], false, false);
            acc[1][j] = __builtin_amdgcn_wmma_f32_16x16x32_f16(false, a1c, false, b, (short)0, acc[1][j], false, false);
        }
        __syncthreads();                  // reads of buf (s&1) done before overwrite

        a0c = a0n; a1c = a1n;
        ap0 += 32; ap1 += 32;
        __builtin_prefetch(ap0 + 32, 0, 1);   // global_prefetch_b8: stage s+2 A line
    }

    // peeled final stage
    wait_async0();
    __syncthreads();
    {
        const _Float16* lb = &sB[(S - 1) & 1][half * 16];
        #pragma unroll
        for (int j = 0; j < 4; ++j) {
            v16h b = *(const v16h*)(lb + (j * 16 + m16) * 32);
            acc[0][j] = __builtin_amdgcn_wmma_f32_16x16x32_f16(false, a0c, false, b, (short)0, acc[0][j], false, false);
            acc[1][j] = __builtin_amdgcn_wmma_f32_16x16x32_f16(false, a1c, false, b, (short)0, acc[1][j], false, false);
        }
    }

    float bv0 = 0.f, bv1 = 0.f, bv2 = 0.f, bv3 = 0.f;
    if (bias) {
        bv0 = bias[col0 +  0 + m16];
        bv1 = bias[col0 + 16 + m16];
        bv2 = bias[col0 + 32 + m16];
        bv3 = bias[col0 + 48 + m16];
    }
    // C/D layout: VGPR r -> row (r + 8*half), lane%16 -> col
    #pragma unroll
    for (int mi = 0; mi < 2; ++mi) {
        #pragma unroll
        for (int r = 0; r < 8; ++r) {
            float* rowp = C + (size_t)(row0 + mi * 16 + half * 8 + r) * ldc + col0 + m16;
            rowp[ 0] = acc[mi][0][r] + bv0;
            rowp[16] = acc[mi][1][r] + bv1;
            rowp[32] = acc[mi][2][r] + bv2;
            rowp[48] = acc[mi][3][r] + bv3;
        }
    }
}

// =====================================================================
// RMSNorm over last dim (768), output f16. One block (256 thr) per row.
// =====================================================================
__global__ __launch_bounds__(256) void k_rmsnorm_f16(
    const float* __restrict__ x, const float* __restrict__ w,
    _Float16* __restrict__ out, int D)
{
    __shared__ float red[256];
    const int m = blockIdx.x;
    const float* xr = x + (size_t)m * D;
    float ss = 0.f;
    for (int i = threadIdx.x; i < D; i += 256) { float v = xr[i]; ss += v * v; }
    red[threadIdx.x] = ss;
    __syncthreads();
    for (int s = 128; s > 0; s >>= 1) {
        if (threadIdx.x < s) red[threadIdx.x] += red[threadIdx.x + s];
        __syncthreads();
    }
    float inv = rsqrtf(red[0] / (float)D + 1e-6f);
    for (int i = threadIdx.x; i < D; i += 256)
        out[(size_t)m * D + i] = (_Float16)(xr[i] * inv * w[i]);
}

// =====================================================================
// conv(K=1) + silu:  xi = silu(xz[:, :ED]*conv_w + conv_b); f32 + f16 out
// =====================================================================
__global__ void k_conv_silu(const float* __restrict__ xz,
                            const float* __restrict__ cw, const float* __restrict__ cb,
                            float* __restrict__ xiF, _Float16* __restrict__ xiH) {
    int t = blockIdx.x * 256 + threadIdx.x;         // < MROWS*ED
    int m = t / ED, e = t % ED;
    float v = xz[(size_t)m * N2 + e] * cw[e] + cb[e];
    float s = v / (1.f + __expf(-v));
    xiF[t] = s;
    xiH[t] = (_Float16)s;
}

// =====================================================================
// dbc post: f16-convert first 48 cols (zero rest), split off Bm, Cm (N=1)
// =====================================================================
__global__ void k_dbc_post(const float* __restrict__ dbc, _Float16* __restrict__ dbcH,
                           float* __restrict__ Bm, float* __restrict__ Cm) {
    int t = blockIdx.x * 256 + threadIdx.x;         // < MROWS*DBCW
    int m = t >> 6, j = t & 63;
    float v = dbc[t];
    dbcH[t] = (j < DTRANK) ? (_Float16)v : (_Float16)0.0f;
    if (j == DTRANK)     Bm[m] = v;
    if (j == DTRANK + 1) Cm[m] = v;
}

// =====================================================================
// Fused selective-scan (N=1): one thread per (b,e) channel, L sequential.
// delta = softplus(dtraw + dt_b); dA = exp(-delta*exp(A_log));
// h = dA*h + delta*xi*Bm;  y = h*Cm + D*xi;  out = y*silu(z) -> f16
// =====================================================================
__global__ void k_scan(const float* __restrict__ dtraw, const float* __restrict__ dtb,
                       const float* __restrict__ Alog, const float* __restrict__ Dp,
                       const float* __restrict__ xiF, const float* __restrict__ Bm,
                       const float* __restrict__ Cm, const float* __restrict__ xz,
                       _Float16* __restrict__ yH) {
    int t = blockIdx.x * 256 + threadIdx.x;         // < BSZ*ED
    int b = t / ED, e = t % ED;
    const float Ac  = -__expf(Alog[e]);             // A_log is (ED, 1)
    const float dpb = dtb[e];
    const float De  = Dp[e];
    float h = 0.f;
    size_t base = (size_t)b * LSEQ;
    for (int l = 0; l < LSEQ; ++l) {
        size_t m   = base + l;
        size_t idx = m * ED + e;
        float dv    = dtraw[idx] + dpb;
        float delta = (dv > 20.f) ? dv : log1pf(__expf(dv));
        float xiv   = xiF[idx];
        float dA    = __expf(delta * Ac);
        h = dA * h + delta * xiv * Bm[m];
        float y = h * Cm[m] + De * xiv;
        float z = xz[m * N2 + ED + e];
        yH[idx] = (_Float16)(y * (z / (1.f + __expf(-z))));
    }
}

// =====================================================================
// Host-side orchestration
// =====================================================================
static inline void gemm(const _Float16* A, int lda, const _Float16* Bt, int ldb,
                        float* C, int ldc, const float* bias, int N, int K,
                        hipStream_t stream) {
    dim3 grid(N / 64, MROWS / 256);
    k_gemm_f16<<<grid, 256, 0, stream>>>(A, lda, Bt, ldb, C, ldc, bias, K);
}

extern "C" void kernel_launch(void* const* d_in, const int* in_sizes, int n_in,
                              void* d_out, int out_size, void* d_ws, size_t ws_size,
                              hipStream_t stream) {
    (void)in_sizes; (void)n_in; (void)out_size; (void)ws_size;

    const int*   tokens   = (const int*)  d_in[0];
    // d_in[1] = n_layers (device scalar) == 8, hardcoded below
    const float* emb_tab  = (const float*)d_in[2];
    const float* emb_proj = (const float*)d_in[3];
    const float* norm_w   = (const float*)d_in[4];
    const float* in_proj  = (const float*)d_in[5];
    const float* conv_w   = (const float*)d_in[6];   // (ED, 1)
    const float* conv_b   = (const float*)d_in[7];
    const float* x_proj   = (const float*)d_in[8];   // (ED, 50)
    const float* dt_proj  = (const float*)d_in[9];   // (48, ED)
    const float* dt_b     = (const float*)d_in[10];
    const float* A_log    = (const float*)d_in[11];  // (ED, 1)
    const float* D_par    = (const float*)d_in[12];
    const float* out_proj = (const float*)d_in[13];  // (ED, D)
    const float* outn_w   = (const float*)d_in[14];
    const float* head_w   = (const float*)d_in[15];  // (D, VOCAB)
    const float* head_b   = (const float*)d_in[16];
    float* out = (float*)d_out;

    char* ws = (char*)d_ws;
    size_t off = 0;
    auto alloc = [&](size_t bytes) -> void* {
        void* p = ws + off;
        off = (off + bytes + 255) & ~(size_t)255;
        return p;
    };

    // f16 transposed weights (N-major, padded)
    _Float16* wtEmb  = (_Float16*)alloc((size_t)DMODEL * EMB    * 2);
    _Float16* wtIn   = (_Float16*)alloc((size_t)N2     * DMODEL * 2);
    _Float16* wtX    = (_Float16*)alloc((size_t)DBCW   * ED     * 2);
    _Float16* wtDt   = (_Float16*)alloc((size_t)ED     * DBCW   * 2);
    _Float16* wtOut  = (_Float16*)alloc((size_t)DMODEL * ED     * 2);
    _Float16* wtHead = (_Float16*)alloc((size_t)VOCAB  * DMODEL * 2);
    // activations
    _Float16* aEmb   = (_Float16*)alloc((size_t)MROWS * EMB    * 2);
    float*    xF     = (float*)   alloc((size_t)MROWS * DMODEL * 4);
    _Float16* hH     = (_Float16*)alloc((size_t)MROWS * DMODEL * 2);
    float*    xzF    = (float*)   alloc((size_t)MROWS * N2     * 4);
    float*    xiF    = (float*)   alloc((size_t)MROWS * ED     * 4);
    _Float16* xiH    = (_Float16*)alloc((size_t)MROWS * ED     * 2);
    float*    dbcF   = (float*)   alloc((size_t)MROWS * DBCW   * 4);
    _Float16* dbcH   = (_Float16*)alloc((size_t)MROWS * DBCW   * 2);
    float*    BmF    = (float*)   alloc((size_t)MROWS * 4);
    float*    CmF    = (float*)   alloc((size_t)MROWS * 4);
    float*    dtraw  = (float*)   alloc((size_t)MROWS * ED * 4);
    _Float16* yH     = (_Float16*)alloc((size_t)MROWS * ED * 2);

    #define TBLK(n) dim3(((n) + 255) / 256), dim3(256)

    // ---- one-time weight prep ----
    k_transpose_w<<<TBLK(DMODEL * EMB),    0, stream>>>(emb_proj, wtEmb,  EMB,    DMODEL, EMB,    DMODEL);
    k_transpose_w<<<TBLK(N2 * DMODEL),     0, stream>>>(in_proj,  wtIn,   DMODEL, N2,     DMODEL, N2);
    k_transpose_w<<<TBLK(DBCW * ED),       0, stream>>>(x_proj,   wtX,    ED,     50,     ED,     DBCW);
    k_transpose_w<<<TBLK(ED * DBCW),       0, stream>>>(dt_proj,  wtDt,   DTRANK, ED,     DBCW,   ED);
    k_transpose_w<<<TBLK(DMODEL * ED),     0, stream>>>(out_proj, wtOut,  ED,     DMODEL, ED,     DMODEL);
    k_transpose_w<<<TBLK(VOCAB * DMODEL),  0, stream>>>(head_w,   wtHead, DMODEL, VOCAB,  DMODEL, VOCAB);

    // ---- embedding: x = emb_table[tokens] @ emb_proj_w ----
    k_embed_gather<<<TBLK(MROWS * EMB), 0, stream>>>(tokens, emb_tab, aEmb);
    gemm(aEmb, EMB, wtEmb, EMB, xF, DMODEL, nullptr, DMODEL, EMB, stream);

    // ---- 8 mamba layers ----
    for (int layer = 0; layer < NLAYER; ++layer) {
        k_rmsnorm_f16<<<MROWS, 256, 0, stream>>>(xF, norm_w, hH, DMODEL);
        gemm(hH, DMODEL, wtIn, DMODEL, xzF, N2, nullptr, N2, DMODEL, stream);
        k_conv_silu<<<TBLK(MROWS * ED), 0, stream>>>(xzF, conv_w, conv_b, xiF, xiH);
        gemm(xiH, ED, wtX, ED, dbcF, DBCW, nullptr, DBCW, ED, stream);
        k_dbc_post<<<TBLK(MROWS * DBCW), 0, stream>>>(dbcF, dbcH, BmF, CmF);
        gemm(dbcH, DBCW, wtDt, DBCW, dtraw, ED, nullptr, ED, DBCW, stream);
        k_scan<<<TBLK(BSZ * ED), 0, stream>>>(dtraw, dt_b, A_log, D_par, xiF, BmF, CmF, xzF, yH);
        gemm(yH, ED, wtOut, ED, xF, DMODEL, nullptr, DMODEL, ED, stream);
    }

    // ---- final norm + LM head ----
    k_rmsnorm_f16<<<MROWS, 256, 0, stream>>>(xF, outn_w, hH, DMODEL);
    gemm(hH, DMODEL, wtHead, DMODEL, out, VOCAB, head_b, VOCAB, DMODEL, stream);
}
